// Attention_49881750176038
// MI455X (gfx1250) — matrix operand order repaired and checked by reading, not compiled, not run on hardware
//
#include <hip/hip_runtime.h>
#include <hip/hip_bf16.h>

typedef __bf16 bf16;
typedef __attribute__((ext_vector_type(16))) __bf16 v16bf;
typedef __attribute__((ext_vector_type(8)))  __bf16 v8bf;
typedef __attribute__((ext_vector_type(4)))  __bf16 v4bf;
typedef __attribute__((ext_vector_type(8)))  float  v8f;

#define BATCH  8
#define SEQ    2048
#define DMODEL 1024
#define DKDIM  512

// ---------------------------------------------------------------------------
// WMMA helpers (gfx1250, wave32): D(16x16,f32) = A(16x32,bf16) x B(32x16,bf16) + C
// A fragment (ISA 7.12.2): lane = 16*hi + r holds row M=r; elements 0..7 are
// K = hi*8 .. hi*8+7, elements 8..15 are K = 16+hi*8 .. +7  -> two b128 loads.
// B fragment (Bt stored row-major [N][K]): lane holds col N=r; same K order as
// A: elements 0..7 K = hi*8 .., elements 8..15 K = 16+hi*8 .. -> two b128 loads.
// C/D: vgpr j, lane 16*hi + r  ->  element (M = j + 8*hi, N = r).
// ---------------------------------------------------------------------------
static __device__ __forceinline__ v16bf ld_frag(const bf16* __restrict__ p, int step) {
  v8bf x = *(const v8bf*)p;
  v8bf y = *(const v8bf*)(p + step);
  v16bf o;
#pragma unroll
  for (int i = 0; i < 8; ++i) { o[i] = x[i]; o[i + 8] = y[i]; }
  return o;
}

static __device__ __forceinline__ v8f wmma_bf16(v16bf a, v16bf b, v8f c) {
  return __builtin_amdgcn_wmma_f32_16x16x32_bf16(false, a, false, b, (short)0, c,
                                                 false, false);
}

// 64(M) x 32(N) wave tile: 8 accumulators, 6 fragment loads (12 b128) per
// K-step of 32 -> 1.5 loads per WMMA. A: row-major [M][K]; Bt: row-major [N][K].
static __device__ __forceinline__ void gemm_wave_tile_64x32(
    const bf16* __restrict__ A0, int lda, const bf16* __restrict__ B0, int ldb,
    int K, int r, int hi, v8f c[4][2]) {
  const bf16* ar = A0 + (size_t)r * lda + hi * 8;
  const bf16* br = B0 + (size_t)r * ldb + hi * 8;
  const size_t a16 = (size_t)16 * lda;
  const size_t b16 = (size_t)16 * ldb;
#pragma unroll 2
  for (int k0 = 0; k0 < K; k0 += 32) {
    v16bf a[4], b[2];
#pragma unroll
    for (int i = 0; i < 4; ++i) a[i] = ld_frag(ar + i * a16 + k0, 16);
#pragma unroll
    for (int j = 0; j < 2; ++j) b[j] = ld_frag(br + j * b16 + k0, 16);
#pragma unroll
    for (int i = 0; i < 4; ++i)
#pragma unroll
      for (int j = 0; j < 2; ++j) c[i][j] = wmma_bf16(a[i], b[j], c[i][j]);
  }
}

// ---------------------------------------------------------------------------
// 1) fp32 -> bf16 convert of activations (query/key/value), blockIdx.z selects.
// ---------------------------------------------------------------------------
__global__ void cvt_act_kernel(const float* __restrict__ q,
                               const float* __restrict__ k,
                               const float* __restrict__ v,
                               bf16* __restrict__ dst) {
  const int z = blockIdx.z;
  const float* src = (z == 0) ? q : ((z == 1) ? k : v);
  size_t idx = ((size_t)blockIdx.x * 256 + threadIdx.x) * 4;
  float4 f = *(const float4*)(src + idx);
  v4bf o = {(bf16)f.x, (bf16)f.y, (bf16)f.z, (bf16)f.w};
  *(v4bf*)(dst + (size_t)z * ((size_t)BATCH * SEQ * DMODEL) + idx) = o;
}

// ---------------------------------------------------------------------------
// 2) fp32 -> bf16 transposed convert of weights: Wb[z][n][k] = W[k][n]
// ---------------------------------------------------------------------------
__global__ void cvt_w_kernel(const float* __restrict__ Wq,
                             const float* __restrict__ Wk,
                             const float* __restrict__ Wv,
                             bf16* __restrict__ dst) {
  const int z = blockIdx.z;
  const float* W = (z == 0) ? Wq : ((z == 1) ? Wk : Wv);
  int idx = blockIdx.x * 256 + threadIdx.x;   // over [n][k], n<512, k<1024
  int n = idx >> 10, kk = idx & 1023;
  dst[(size_t)z * (DKDIM * DMODEL) + idx] = (bf16)W[(size_t)kk * DKDIM + n];
}

// ---------------------------------------------------------------------------
// 3) Projection GEMM: [16384,1024]bf16 x Wt[512,1024] -> q/k bf16 (row major),
//    v stored transposed per batch: vbT[b][n][s].
//    Block tile 128x128 (8 waves, 2x4 of 64x32).  grid (128, 4, 3), 256 thr.
// ---------------------------------------------------------------------------
__global__ void proj_gemm_kernel(const bf16* __restrict__ Xb,
                                 const bf16* __restrict__ Wb,
                                 bf16* __restrict__ qb, bf16* __restrict__ kb,
                                 bf16* __restrict__ vbT) {
  const int z = blockIdx.z;
  const bf16* A = Xb + (size_t)z * ((size_t)BATCH * SEQ * DMODEL);
  const bf16* Bt = Wb + (size_t)z * (DKDIM * DMODEL);
  const int lane = threadIdx.x & 31, wave = threadIdx.x >> 5;
  const int r = lane & 15, hi = lane >> 4;
  const int bm = blockIdx.x * 128 + (wave & 1) * 64;
  const int bn = blockIdx.y * 128 + (wave >> 1) * 32;
  v8f c[4][2] = {};
  gemm_wave_tile_64x32(A + (size_t)bm * DMODEL, DMODEL,
                       Bt + (size_t)bn * DMODEL, DMODEL, DMODEL, r, hi, c);
  if (z == 2) {
#pragma unroll
    for (int ti = 0; ti < 4; ++ti)
#pragma unroll
      for (int tj = 0; tj < 2; ++tj)
#pragma unroll
        for (int j = 0; j < 8; ++j) {
          int row = bm + ti * 16 + hi * 8 + j;   // global s index (b*2048 + s)
          int col = bn + tj * 16 + r;            // dk index
          int b = row >> 11, s = row & 2047;
          vbT[((size_t)b * DKDIM + col) * SEQ + s] = (bf16)c[ti][tj][j];
        }
  } else {
    bf16* D = (z == 0) ? qb : kb;
#pragma unroll
    for (int ti = 0; ti < 4; ++ti)
#pragma unroll
      for (int tj = 0; tj < 2; ++tj)
#pragma unroll
        for (int j = 0; j < 8; ++j) {
          int row = bm + ti * 16 + hi * 8 + j;
          int col = bn + tj * 16 + r;
          D[(size_t)row * DKDIM + col] = (bf16)c[ti][tj][j];
        }
  }
}

// ---------------------------------------------------------------------------
// 4) Scores: pre[b][q][s] = (qb[b] @ kb[b]^T)/sqrt(512) - 1e9*mask.
//    Mask is a pure 134MB stream -> non-temporal loads (keep q/k/v in L2).
//    grid (16, 16, 8), 256 thr.
// ---------------------------------------------------------------------------
__global__ void score_gemm_kernel(const bf16* __restrict__ qb,
                                  const bf16* __restrict__ kb,
                                  const float* __restrict__ mask,
                                  float* __restrict__ score) {
  const int b = blockIdx.z;
  const bf16* A = qb + (size_t)b * SEQ * DKDIM;
  const bf16* Bt = kb + (size_t)b * SEQ * DKDIM;
  float* D = score + (size_t)b * SEQ * SEQ;
  const float* Mk = mask + (size_t)b * SEQ * SEQ;
  const int lane = threadIdx.x & 31, wave = threadIdx.x >> 5;
  const int r = lane & 15, hi = lane >> 4;
  const int bm = blockIdx.x * 128 + (wave & 1) * 64;
  const int bn = blockIdx.y * 128 + (wave >> 1) * 32;
  v8f c[4][2] = {};
  gemm_wave_tile_64x32(A + (size_t)bm * DKDIM, DKDIM, Bt + (size_t)bn * DKDIM,
                       DKDIM, DKDIM, r, hi, c);
  const float scale = 0.044194173824159216f;  // 1/sqrt(512)
#pragma unroll
  for (int ti = 0; ti < 4; ++ti)
#pragma unroll
    for (int tj = 0; tj < 2; ++tj)
#pragma unroll
      for (int j = 0; j < 8; ++j) {
        int row = bm + ti * 16 + hi * 8 + j;
        int col = bn + tj * 16 + r;
        size_t o = (size_t)row * SEQ + col;
        float m = __builtin_nontemporal_load(Mk + o);
        D[o] = c[ti][tj][j] * scale + m * (-1e9f);
      }
}

// ---------------------------------------------------------------------------
// 5) Softmax over last axis. Final fp32 scores are never re-read -> NT stores;
//    bf16 copy of P feeds the PV GEMM (keep cached).
//    grid (16384), 256 thr (8 waves), 8 elems/thread.
// ---------------------------------------------------------------------------
__global__ void softmax_kernel(float* __restrict__ score, bf16* __restrict__ Pb) {
  const int row = blockIdx.x;
  const int tid = threadIdx.x;
  float* srow = score + (size_t)row * SEQ;
  bf16* prow = Pb + (size_t)row * SEQ;
  __shared__ float red_m[8];
  __shared__ float red_s[8];
  float v[8];
  float m = -3.4e38f;
#pragma unroll
  for (int i = 0; i < 8; ++i) {
    v[i] = srow[tid + 256 * i];
    m = fmaxf(m, v[i]);
  }
#pragma unroll
  for (int off = 16; off; off >>= 1) m = fmaxf(m, __shfl_xor(m, off, 32));
  if ((tid & 31) == 0) red_m[tid >> 5] = m;
  __syncthreads();
#pragma unroll
  for (int i = 0; i < 8; ++i) m = fmaxf(m, red_m[i]);
  float s = 0.f;
#pragma unroll
  for (int i = 0; i < 8; ++i) {
    v[i] = __expf(v[i] - m);
    s += v[i];
  }
#pragma unroll
  for (int off = 16; off; off >>= 1) s += __shfl_xor(s, off, 32);
  if ((tid & 31) == 0) red_s[tid >> 5] = s;
  __syncthreads();
  s = 0.f;
#pragma unroll
  for (int i = 0; i < 8; ++i) s += red_s[i];
  float inv = 1.0f / s;
#pragma unroll
  for (int i = 0; i < 8; ++i) {
    float p = v[i] * inv;
    __builtin_nontemporal_store(p, srow + tid + 256 * i);
    prow[tid + 256 * i] = (bf16)p;
  }
}

// ---------------------------------------------------------------------------
// 6) out[b] = P[b] @ v[b] : [2048,2048]bf16 x vbT[b]([512,2048], Bt layout).
//    Output never re-read -> NT stores.  grid (16, 4, 8), 256 thr.
// ---------------------------------------------------------------------------
__global__ void out_gemm_kernel(const bf16* __restrict__ Pb,
                                const bf16* __restrict__ vbT,
                                float* __restrict__ out) {
  const int b = blockIdx.z;
  const bf16* A = Pb + (size_t)b * SEQ * SEQ;
  const bf16* Bt = vbT + (size_t)b * DKDIM * SEQ;
  float* D = out + (size_t)b * SEQ * DKDIM;
  const int lane = threadIdx.x & 31, wave = threadIdx.x >> 5;
  const int r = lane & 15, hi = lane >> 4;
  const int bm = blockIdx.x * 128 + (wave & 1) * 64;
  const int bn = blockIdx.y * 128 + (wave >> 1) * 32;
  v8f c[4][2] = {};
  gemm_wave_tile_64x32(A + (size_t)bm * SEQ, SEQ, Bt + (size_t)bn * SEQ, SEQ,
                       SEQ, r, hi, c);
#pragma unroll
  for (int ti = 0; ti < 4; ++ti)
#pragma unroll
    for (int tj = 0; tj < 2; ++tj)
#pragma unroll
      for (int j = 0; j < 8; ++j) {
        int row = bm + ti * 16 + hi * 8 + j;
        int col = bn + tj * 16 + r;
        __builtin_nontemporal_store(c[ti][tj][j], D + (size_t)row * DKDIM + col);
      }
}

// ---------------------------------------------------------------------------
// Launch. Workspace layout (bytes):
//   [0, 100663296)            : Xbf16 (3 x 16384x1024 bf16)  -- later reused as Pbf16
//   [100663296, 103809024)    : Wbf16 transposed (3 x 512x1024 bf16)
//   [103809024, 154140672)    : qb, kb, vbT (3 x 16384x512 bf16)
// ---------------------------------------------------------------------------
extern "C" void kernel_launch(void* const* d_in, const int* in_sizes, int n_in,
                              void* d_out, int out_size, void* d_ws,
                              size_t ws_size, hipStream_t stream) {
  const float* query = (const float*)d_in[0];
  const float* key_  = (const float*)d_in[1];
  const float* value = (const float*)d_in[2];
  const float* mask  = (const float*)d_in[3];
  const float* Wq    = (const float*)d_in[4];
  const float* Wk    = (const float*)d_in[5];
  const float* Wv    = (const float*)d_in[6];

  float* out   = (float*)d_out;                         // [8,2048,512]
  float* score = out + (size_t)BATCH * SEQ * DKDIM;     // [8,2048,2048]

  char* ws = (char*)d_ws;
  bf16* Xb  = (bf16*)ws;
  bf16* Wb  = (bf16*)(ws + 100663296ull);
  bf16* qb  = (bf16*)(ws + 103809024ull);
  bf16* kb  = qb + (size_t)BATCH * SEQ * DKDIM;
  bf16* vbT = kb + (size_t)BATCH * SEQ * DKDIM;
  bf16* Pb  = (bf16*)ws;  // reuse Xb region after projections complete

  cvt_act_kernel<<<dim3(16384, 1, 3), 256, 0, stream>>>(query, key_, value, Xb);
  cvt_w_kernel<<<dim3(2048, 1, 3), 256, 0, stream>>>(Wq, Wk, Wv, Wb);
  proj_gemm_kernel<<<dim3(128, 4, 3), 256, 0, stream>>>(Xb, Wb, qb, kb, vbT);
  score_gemm_kernel<<<dim3(16, 16, 8), 256, 0, stream>>>(qb, kb, mask, score);
  softmax_kernel<<<dim3(16384, 1, 1), 256, 0, stream>>>(score, Pb);
  out_gemm_kernel<<<dim3(16, 4, 8), 256, 0, stream>>>(Pb, vbT, out);
}
